// CrossAttention_2430951490253
// MI455X (gfx1250) — matrix operand order repaired and checked by reading, not compiled, not run on hardware
//
#include <hip/hip_runtime.h>

#if defined(__has_builtin)
#  if __has_builtin(__builtin_amdgcn_tensor_load_to_lds) && __has_builtin(__builtin_amdgcn_s_wait_tensorcnt)
#    define HAVE_TDM 1
#  else
#    define HAVE_TDM 0
#  endif
#else
#  define HAVE_TDM 0
#endif

typedef __bf16 bf16;
typedef __attribute__((ext_vector_type(16))) __bf16 v16bf;
typedef __attribute__((ext_vector_type(8)))  float  v8f;
typedef __attribute__((ext_vector_type(4)))  unsigned int u32x4;
typedef __attribute__((ext_vector_type(8)))  int i32x8;
typedef __attribute__((ext_vector_type(4)))  int i32x4;

namespace {
constexpr int Bc = 16;          // batch
constexpr int Nq = 4096;        // query length
constexpr int Mc = 77;          // context length
constexpr int QD = 1024;        // query dim
constexpr int CD = 768;         // context dim
constexpr int Hn = 8;           // heads
constexpr int Dh = 64;          // head dim
constexpr int IN = 512;         // inner = H*D
constexpr int MROWS = Bc * Nq;  // 65536
constexpr int CROWS = Bc * Mc;  // 1232
constexpr int MP = 96;          // context length padded for WMMA
constexpr bool kHaveTDM = (HAVE_TDM != 0);
}

union V16U { uint4 q[2]; v16bf v; };
union V8HU { uint4 q; bf16 h[8]; };

__device__ inline v16bf ld16(const bf16* p) {          // 16 contiguous, 16B aligned
  V16U t; t.q[0] = ((const uint4*)p)[0]; t.q[1] = ((const uint4*)p)[1]; return t.v;
}
__device__ inline v16bf ld8x2(const bf16* p0, const bf16* p1) { // two aligned groups of 8
  V16U t; t.q[0] = ((const uint4*)p0)[0]; t.q[1] = ((const uint4*)p1)[0]; return t.v;
}
__device__ inline v8f wmma_bf16(v16bf a, v16bf b, v8f c) {
  return __builtin_amdgcn_wmma_f32_16x16x32_bf16(false, a, false, b, (short)0, c, false, false);
}
__device__ inline unsigned lds_off(const void* p) {    // LDS aperture: low 32 bits = offset
  return (unsigned)(size_t)p;
}

// ---------- Tensor Data Mover: 2-D bf16 tile load, global -> LDS ----------
// D# per CDNA5 ISA ch.8: group0 = {flags, lds_addr, gaddr_lo, gaddr_hi|type},
// group1 packs data_size/pad + tensor_dim0/1 + tile_dim0/1 + stride0.
// OOB rows (y >= tdim1) zero-fill; pad_amount DWORDs inserted in LDS after every
// pad_interval DWORDs to preserve the padded row pitch.
__device__ inline void tdm_load_2d_bf16(unsigned lds_byte, unsigned long long gaddr,
                                        unsigned tdim0, unsigned tdim1,
                                        unsigned tile0, unsigned tile1,
                                        unsigned stride0,
                                        unsigned pad_int_code, unsigned pad_amt_code) {
#if HAVE_TDM
  u32x4 g0;
  g0[0] = 1u;                                                  // count=1 user D#
  g0[1] = lds_byte;                                            // lds_addr
  g0[2] = (unsigned)(gaddr & 0xffffffffull);                   // global_addr[31:0]
  g0[3] = (unsigned)((gaddr >> 32) & 0x01ffffffull) | (2u << 30); // [56:32] | type=2
  i32x8 g1;
  unsigned w0 = (1u << 16)                                      // data_size = 2 bytes
              | (1u << 20)                                      // pad_enable
              | (pad_int_code << 22) | (pad_amt_code << 25);
  g1[0] = (int)w0;
  g1[1] = (int)(tdim0 << 16);                                   // tensor_dim0[15:0] -> [31:16]
  g1[2] = (int)((tdim0 >> 16) | (tdim1 << 16));                 // dim0 hi | dim1 lo
  g1[3] = (int)((tdim1 >> 16) | (tile0 << 16));                 // dim1 hi | tile_dim0
  g1[4] = (int)(tile1 & 0xffffu);                               // tile_dim1 (tile_dim2=0)
  g1[5] = (int)stride0;                                         // tensor_dim0_stride lo32
  g1[6] = 0;                                                    // stride0 hi16 | stride1 lo
  g1[7] = 0;
  i32x4 z4 = {0, 0, 0, 0};                                      // groups 2/3: beyond-2D unused
  i32x8 z8i = {0, 0, 0, 0, 0, 0, 0, 0};
  __builtin_amdgcn_tensor_load_to_lds(g0, g1, z4, z4, z8i, 0);
#else
  (void)lds_byte; (void)gaddr; (void)tdim0; (void)tdim1; (void)tile0; (void)tile1;
  (void)stride0; (void)pad_int_code; (void)pad_amt_code;
#endif
}
__device__ inline void tdm_wait() {
#if HAVE_TDM
  __builtin_amdgcn_s_wait_tensorcnt(0);
#endif
}

// ---------------- weight convert + transpose: Wt[n][k] = (bf16)W[k][n] ----------------
__global__ void cvt_transpose(const float* __restrict__ W, bf16* __restrict__ Wt, int K, int N) {
  int idx = blockIdx.x * blockDim.x + threadIdx.x;
  if (idx < K * N) {
    int n = idx / K, k = idx - n * K;     // writes coalesced over k
    Wt[idx] = (bf16)W[(size_t)k * N + n];
  }
}

// ---------------- A-tile register stage (f32 converts on the fly) ----------------
template <typename T> struct ALoad;
template <> struct ALoad<float> {
  float4 f0, f1;
  __device__ inline void fetch(const float* src, bool valid) {
    f0 = make_float4(0.f,0.f,0.f,0.f); f1 = f0;
    if (valid) { f0 = ((const float4*)src)[0]; f1 = ((const float4*)src)[1]; }
  }
  __device__ inline void commit(bf16* dst) const {
    V8HU t;
    t.h[0]=(bf16)f0.x; t.h[1]=(bf16)f0.y; t.h[2]=(bf16)f0.z; t.h[3]=(bf16)f0.w;
    t.h[4]=(bf16)f1.x; t.h[5]=(bf16)f1.y; t.h[6]=(bf16)f1.z; t.h[7]=(bf16)f1.w;
    *((uint4*)dst) = t.q;
  }
};
template <> struct ALoad<bf16> {
  uint4 u;
  __device__ inline void fetch(const bf16* src, bool valid) {
    u = make_uint4(0u,0u,0u,0u);
    if (valid) u = ((const uint4*)src)[0];
  }
  __device__ inline void commit(bf16* dst) const { *((uint4*)dst) = u; }
};

// ---------------- tiled WMMA GEMM: C[M,N] = A[M,K] @ Bt[N,K]^T (+bias) ----------------
// 256 threads = 8 waves (2x4); block tile 64x256, BK=32, wave tile 32x64.
// Double-buffered LDS; bf16 A-tiles optionally DMA'd by the Tensor Data Mover.
template <typename AT, bool BIAS, bool OUT_BF16>
__global__ __launch_bounds__(256)
void gemm_wmma(const AT* __restrict__ A, const bf16* __restrict__ Bt,
               void* __restrict__ Cout, const float* __restrict__ bias,
               int M, int N, int K) {
  __shared__ bf16 lA[2][64][40];    // [m][k], 80B row pitch (16 DW + 4 DW pad)
  __shared__ bf16 lB[2][256][40];   // [n][k], same pitch

  constexpr bool TDM_A = kHaveTDM && (sizeof(AT) == 2);

  const int tid  = threadIdx.x;
  const int wid  = tid >> 5;
  const int lane = tid & 31;
  const int lrow = lane & 15;
  const int lhi  = lane >> 4;
  const int waveM = wid >> 2;       // 0..1
  const int waveN = wid & 3;        // 0..3

  const int arow = tid >> 2;        // 0..63
  const int acol = (tid & 3) * 8;   // 0,8,16,24
  const int grow0 = blockIdx.y * 64;
  const int grow = grow0 + arow;
  const bool avalid = grow < M;

  const bf16* BtRow = Bt + (size_t)(blockIdx.x * 256 + tid) * K;  // one tile row per thread

  v8f acc[2][4];
  const v8f z8 = {0.f,0.f,0.f,0.f,0.f,0.f,0.f,0.f};
#pragma unroll
  for (int mi = 0; mi < 2; ++mi)
#pragma unroll
    for (int ni = 0; ni < 4; ++ni) acc[mi][ni] = z8;

  ALoad<AT> ald;
  uint4 breg[4];

  // ---- prologue: tile 0 -> LDS[0] ----
  if (TDM_A) {
    if (wid == 0)
      tdm_load_2d_bf16(lds_off(&lA[0][0][0]),
                       (unsigned long long)(size_t)(A + (size_t)grow0 * K),
                       /*tdim0*/32, /*tdim1*/(unsigned)(M - grow0),
                       /*tile*/32, 64, (unsigned)K, /*pad 16DW*/3, /*+4DW*/3);
  } else {
    ald.fetch(A + (size_t)grow * K + acol, avalid);
  }
#pragma unroll
  for (int p = 0; p < 4; ++p) breg[p] = ((const uint4*)(BtRow + p * 8))[0];
  if (!TDM_A) ald.commit(&lA[0][arow][acol]);
#pragma unroll
  for (int p = 0; p < 4; ++p) *((uint4*)&lB[0][tid][p * 8]) = breg[p];
  if (TDM_A && wid == 0) tdm_wait();
  __syncthreads();

  const int nk = K >> 5;
  for (int it = 0; it < nk; ++it) {
    const int buf = it & 1, nbuf = buf ^ 1;
    const int k0n = (it + 1) * 32;
    const bool hasNext = (it + 1) < nk;

    if (hasNext) {                                   // stage tile it+1 (overlaps WMMA)
      if (TDM_A) {
        if (wid == 0)
          tdm_load_2d_bf16(lds_off(&lA[nbuf][0][0]),
                           (unsigned long long)(size_t)(A + (size_t)grow0 * K + k0n),
                           32, (unsigned)(M - grow0), 32, 64, (unsigned)K, 3, 3);
      } else {
        ald.fetch(A + (size_t)grow * K + k0n + acol, avalid);
        __builtin_prefetch(A + (size_t)grow * K + k0n + 32 + acol, 0, 1);
      }
#pragma unroll
      for (int p = 0; p < 4; ++p) breg[p] = ((const uint4*)(BtRow + k0n + p * 8))[0];
    }

    // ---- compute on LDS[buf] ----
    v16bf afrag[2];
#pragma unroll
    for (int mi = 0; mi < 2; ++mi) {
      const bf16* ar = &lA[buf][waveM * 32 + mi * 16 + lrow][0];
      afrag[mi] = ld8x2(ar + lhi * 8, ar + 16 + lhi * 8);
    }
    v16bf bfrag[4];
#pragma unroll
    for (int ni = 0; ni < 4; ++ni) {
      const bf16* br = &lB[buf][waveN * 64 + ni * 16 + lrow][0];
      bfrag[ni] = ld8x2(br + lhi * 8, br + 16 + lhi * 8);
    }
#pragma unroll
    for (int mi = 0; mi < 2; ++mi)
#pragma unroll
      for (int ni = 0; ni < 4; ++ni)
        acc[mi][ni] = wmma_bf16(afrag[mi], bfrag[ni], acc[mi][ni]);

    if (hasNext) {
      if (!TDM_A) ald.commit(&lA[nbuf][arow][acol]);
#pragma unroll
      for (int p = 0; p < 4; ++p) *((uint4*)&lB[nbuf][tid][p * 8]) = breg[p];
      if (TDM_A && wid == 0) tdm_wait();
    }
    __syncthreads();
  }

#pragma unroll
  for (int mi = 0; mi < 2; ++mi)
#pragma unroll
    for (int ni = 0; ni < 4; ++ni)
#pragma unroll
      for (int r = 0; r < 8; ++r) {
        int gm = grow0 + waveM * 32 + mi * 16 + lhi * 8 + r;
        int gn = blockIdx.x * 256 + waveN * 64 + ni * 16 + lrow;
        if (gm < M) {
          float val = acc[mi][ni][r];
          if (BIAS) val += bias[gn];
          if (OUT_BF16) ((bf16*)Cout)[(size_t)gm * N + gn] = (bf16)val;
          else          ((float*)Cout)[(size_t)gm * N + gn] = val;
        }
      }
}

// ---------------- fused attention: softmax(Q K^T / 8) V per (b,h) ----------------
// grid: (Nq/128, B*H); 256 threads = 8 waves, 16 query rows per wave
__global__ __launch_bounds__(256)
void attn_wmma(const bf16* __restrict__ q, const bf16* __restrict__ k,
               const bf16* __restrict__ v, bf16* __restrict__ o) {
  __shared__ bf16 kl[MP][72];        // K:  [m][d]   (128B row + 16B pad)
  __shared__ bf16 vl[Dh][104];       // V^T:[d][m]
  __shared__ bf16 pl[8][16][104];    // P per wave

  const int bh = blockIdx.y;
  const int b  = bh / Hn;
  const int h  = bh % Hn;
  const int tid  = threadIdx.x;
  const int wid  = tid >> 5;
  const int lane = tid & 31;
  const int lrow = lane & 15;
  const int lhi  = lane >> 4;

  // ---- K/V staging; K tile via Tensor Data Mover when available ----
  if (kHaveTDM) {
    if (wid == 0) {
      tdm_load_2d_bf16(lds_off(&kl[0][0]),
                       (unsigned long long)(size_t)(k + (size_t)(b * Mc) * IN + h * Dh),
                       /*tdim0*/Dh, /*tdim1*/Mc,          // rows >= 77 zero-fill
                       /*tile*/Dh, MP, (unsigned)IN,
                       /*pad after 32 DW*/4, /*+4 DW*/3);
      tdm_wait();
    }
    for (int idx = tid; idx < MP * Dh; idx += 256) {
      int m = idx >> 6, d = idx & 63;
      vl[d][m] = (m < Mc) ? v[(size_t)(b * Mc + m) * IN + h * Dh + d] : (bf16)0.f;
    }
  } else {
    for (int idx = tid; idx < MP * Dh; idx += 256) {
      int m = idx >> 6, d = idx & 63;
      bf16 kvv = (bf16)0.f, vvv = (bf16)0.f;
      if (m < Mc) {
        size_t base = (size_t)(b * Mc + m) * IN + h * Dh + d;
        kvv = k[base];
        vvv = v[base];
      }
      kl[m][d] = kvv;
      vl[d][m] = vvv;
    }
  }
  __syncthreads();

  const int n0 = blockIdx.x * 128 + wid * 16;

  // ---- scores S = Q K^T : 6 column fragments x 2 K-steps ----
  v8f sacc[6];
  const v8f z8 = {0.f,0.f,0.f,0.f,0.f,0.f,0.f,0.f};
#pragma unroll
  for (int f = 0; f < 6; ++f) sacc[f] = z8;

#pragma unroll
  for (int ds = 0; ds < 64; ds += 32) {
    const bf16* qp = q + (size_t)(b * Nq + n0 + lrow) * IN + h * Dh + ds;
    v16bf aq = ld8x2(qp + lhi * 8, qp + 16 + lhi * 8);
#pragma unroll
    for (int f = 0; f < 6; ++f) {
      v16bf bk = ld8x2(&kl[f * 16 + lrow][ds + lhi * 8], &kl[f * 16 + lrow][ds + 16 + lhi * 8]);
      sacc[f] = wmma_bf16(aq, bk, sacc[f]);
    }
  }

  // ---- softmax over 96 cols (77 valid); rows split per 16-lane half ----
  const float scale = 0.125f;   // D^-0.5
#pragma unroll
  for (int r = 0; r < 8; ++r) {
    float mx = -3.0e38f;
    float sv[6];
#pragma unroll
    for (int f = 0; f < 6; ++f) {
      int col = f * 16 + lrow;
      float s = sacc[f][r] * scale;
      sv[f] = s;
      if (col < Mc) mx = fmaxf(mx, s);
    }
#pragma unroll
    for (int off = 1; off < 16; off <<= 1) mx = fmaxf(mx, __shfl_xor(mx, off, 32));
    float sum = 0.f;
    float pv[6];
#pragma unroll
    for (int f = 0; f < 6; ++f) {
      int col = f * 16 + lrow;
      float p = (col < Mc) ? __expf(sv[f] - mx) : 0.f;
      pv[f] = p;
      sum += p;
    }
#pragma unroll
    for (int off = 1; off < 16; off <<= 1) sum += __shfl_xor(sum, off, 32);
    float rinv = 1.f / sum;
#pragma unroll
    for (int f = 0; f < 6; ++f)
      pl[wid][lhi * 8 + r][f * 16 + lrow] = (bf16)(pv[f] * rinv);
  }

  // ---- O = P V : 4 column fragments x 3 K-steps over m ----
  v8f oacc[4];
#pragma unroll
  for (int ni = 0; ni < 4; ++ni) oacc[ni] = z8;
#pragma unroll
  for (int ks = 0; ks < 3; ++ks) {
    int kb = ks * 32;
    const bf16* pr = &pl[wid][lrow][kb];
    v16bf ap = ld8x2(pr + lhi * 8, pr + 16 + lhi * 8);
#pragma unroll
    for (int ni = 0; ni < 4; ++ni) {
      v16bf bv = ld8x2(&vl[ni * 16 + lrow][kb + lhi * 8], &vl[ni * 16 + lrow][kb + 16 + lhi * 8]);
      oacc[ni] = wmma_bf16(ap, bv, oacc[ni]);
    }
  }

#pragma unroll
  for (int ni = 0; ni < 4; ++ni)
#pragma unroll
    for (int r = 0; r < 8; ++r) {
      int gr = n0 + lhi * 8 + r;
      int d  = ni * 16 + lrow;
      o[(size_t)(b * Nq + gr) * IN + h * Dh + d] = (bf16)oacc[ni][r];
    }
}

// ---------------- host launch ----------------
extern "C" void kernel_launch(void* const* d_in, const int* in_sizes, int n_in,
                              void* d_out, int out_size, void* d_ws, size_t ws_size,
                              hipStream_t stream) {
  (void)in_sizes; (void)n_in; (void)out_size; (void)ws_size;
  const float* x   = (const float*)d_in[0];
  const float* ctx = (const float*)d_in[1];
  const float* Wq  = (const float*)d_in[2];
  const float* Wk  = (const float*)d_in[3];
  const float* Wv  = (const float*)d_in[4];
  const float* Wo  = (const float*)d_in[5];
  const float* bo  = (const float*)d_in[6];
  float* out = (float*)d_out;

  bf16* p = (bf16*)d_ws;
  bf16* WqT = p; p += (size_t)QD * IN;   // [IN][QD]
  bf16* WkT = p; p += (size_t)CD * IN;   // [IN][CD]
  bf16* WvT = p; p += (size_t)CD * IN;   // [IN][CD]
  bf16* WoT = p; p += (size_t)IN * QD;   // [QD][IN]
  bf16* qb  = p; p += (size_t)MROWS * IN;
  bf16* kb  = p; p += (size_t)CROWS * IN;
  bf16* vb  = p; p += (size_t)CROWS * IN;
  bf16* ob  = p; p += (size_t)MROWS * IN;

  // weight convert + transpose (tiny, once)
  cvt_transpose<<<(QD * IN + 255) / 256, 256, 0, stream>>>(Wq, WqT, QD, IN);
  cvt_transpose<<<(CD * IN + 255) / 256, 256, 0, stream>>>(Wk, WkT, CD, IN);
  cvt_transpose<<<(CD * IN + 255) / 256, 256, 0, stream>>>(Wv, WvT, CD, IN);
  cvt_transpose<<<(IN * QD + 255) / 256, 256, 0, stream>>>(Wo, WoT, IN, QD);

  // q = x @ Wq   (65536 x 1024 x 512), f32 A converted on the fly
  gemm_wmma<float, false, true><<<dim3(IN / 256, MROWS / 64), 256, 0, stream>>>(
      x, WqT, (void*)qb, nullptr, MROWS, IN, QD);
  // k/v = context @ Wk/Wv  (1232 x 768 x 512)
  gemm_wmma<float, false, true><<<dim3(IN / 256, (CROWS + 63) / 64), 256, 0, stream>>>(
      ctx, WkT, (void*)kb, nullptr, CROWS, IN, CD);
  gemm_wmma<float, false, true><<<dim3(IN / 256, (CROWS + 63) / 64), 256, 0, stream>>>(
      ctx, WvT, (void*)vb, nullptr, CROWS, IN, CD);

  // fused attention
  attn_wmma<<<dim3(Nq / 128, Bc * Hn), 256, 0, stream>>>(qb, kb, vb, ob);

  // out = attn_out @ Wo + bo  (65536 x 512 x 1024), bf16 A staged by TDM, f32 out
  gemm_wmma<bf16, true, false><<<dim3(QD / 256, MROWS / 64), 256, 0, stream>>>(
      ob, WoT, (void*)out, bo, MROWS, QD, IN);
}
